// AudioMultiResCRNNFiLM_47545287967314
// MI455X (gfx1250) — matrix operand order repaired and checked by reading, not compiled, not run on hardware
//
#include <hip/hip_runtime.h>
#include <math.h>

typedef __attribute__((ext_vector_type(16))) _Float16 v16h;
typedef __attribute__((ext_vector_type(8)))  _Float16 v8h;
typedef __attribute__((ext_vector_type(8)))  float    v8f;

#define BB     16
#define NMELS  128
#define TT     512
#define GHC    256
#define T2     32
#define BN_INVF 0.9999950000374997f   /* 1/sqrt(1+1e-5) */
#define S_VALF  (320.0f/(32000.0f*0.06f))

static __device__ __forceinline__ v16h cat8(v8h a, v8h b) {
  return __builtin_shufflevector(a, b, 0,1,2,3,4,5,6,7,8,9,10,11,12,13,14,15);
}
/* floor(a/b) for small non-negative ints via v_rcp_f32.
 * Exact: quotient <= 4608, rel err of rcp ~2^-23, +0.5/b bias keeps the
 * result strictly inside (q, q+1) for every representable case here.     */
static __device__ __forceinline__ int fdiv_floor(int a, int b) {
  return (int)(((float)a + 0.5f) * __builtin_amdgcn_rcpf((float)b));
}

/* ---------------- weight f32 -> f16 (with K padding to mult of 32) ------- */
__global__ void k_w2h(const float* __restrict__ src, _Float16* __restrict__ dst,
                      int N, int Kt, int Kp) {
  long i = (long)blockIdx.x * 256 + threadIdx.x;
  long tot = (long)N * Kp;
  if (i >= tot) return;
  int n = (int)(i / Kp), k = (int)(i % Kp);
  float v = (k < Kt) ? src[(long)n * Kt + k] : 0.f;
  dst[i] = (_Float16)v;
}

/* ---------------- PCEN + input BN --------------------------------------- */
__global__ void k_pcen(const float* __restrict__ mel_big, const float* __restrict__ mel_small,
                       const float* pa0, const float* pd0, const float* pr0,
                       const float* pa1, const float* pd1, const float* pr1,
                       const float* ibn_b, const float* ibn_g, float* __restrict__ spec) {
  int row = blockIdx.x * blockDim.x + threadIdx.x;
  if (row >= 2 * BB * NMELS) return;
  int s = row / (BB * NMELS);
  int rem = row % (BB * NMELS);
  int b = rem / NMELS, f = rem % NMELS;
  const float* mel = (s == 0 ? mel_big : mel_small) + ((long)(b * NMELS + f)) * TT;
  float alpha = (s == 0 ? pa0[0] : pa1[0]);
  alpha = fminf(fmaxf(alpha, 0.01f), 0.99f);
  float delta = fabsf(s == 0 ? pd0[0] : pd1[0]) + 1e-6f;
  float r = (s == 0 ? pr0[0] : pr1[0]);
  r = fminf(fmaxf(r, 0.01f), 1.0f);
  float dr = powf(delta, r);
  float sc = ibn_g[s] * BN_INVF, bi = ibn_b[s];
  float* op = spec + (((long)(b * 2 + s)) * NMELS + f) * TT;
  float M = mel[0];
  for (int t = 0; t < TT; ++t) {
    float x = mel[t];
    if (t) M = (1.0f - S_VALF) * M + S_VALF * x;
    float smooth = powf(1e-6f + M, alpha);
    float pv = powf(x / (smooth + 1e-6f) + delta, r) - dr;
    op[t] = pv * sc + bi;
  }
}

/* ---------------- generic implicit-GEMM / GEMM with WMMA ------------------
 * C[M,N] = A[M,K] * W[N,K]^T. Block tile 64Mx64N, 8 waves, each wave 16x32
 * (2 WMMAs per K-chunk). A tile in LDS row-major [m][k] (pad 40), B tile
 * fragment-major [n][k] (pad 40) so fragments are ds_load_b128 pairs.
 * conv_mode=1: A is im2col built on the fly (row decode hoisted out of the
 * K loop, k decode via v_rcp-based floor division).
 * epi: 1 = NCHW out with BN(scale,shift)+optional residual+optional relu
 *      0 = row-major out + bias ; 2 = tanh(row-major out + bias)          */
__global__ __launch_bounds__(256) void k_wmma_gemm(
    const float* __restrict__ X, const _Float16* __restrict__ W, float* __restrict__ out,
    int M, int N, int Ktrue, int Kpad,
    int conv_mode, int Cin, int Hin, int Win, int KH, int KW, int stride, int pad,
    int HO, int WO,
    int epi, const float* __restrict__ scale, const float* __restrict__ shift,
    const float* __restrict__ resid, int relu) {
  __shared__ alignas(16) _Float16 Alds[64][40];
  __shared__ alignas(16) _Float16 Blds[64][40];
  int tid = threadIdx.x, lane = tid & 31, wave = tid >> 5;
  int wm = wave & 3, wn2 = wave >> 2;
  int m0 = blockIdx.x * 64, n0 = blockIdx.y * 64;
  int khkw = KH * KW, howo = HO * WO;

  /* hoisted per-thread row decode for the conv A-loader (rows wave+8*it) */
  const float* xb[8]; int h0a[8], w0a[8]; bool mok[8];
  if (conv_mode) {
#pragma unroll
    for (int it = 0; it < 8; ++it) {
      int m = m0 + wave + 8 * it;
      mok[it] = (m < M);
      int mm = mok[it] ? m : 0;
      int b = fdiv_floor(mm, howo);
      int rm = mm - b * howo;
      int ho = fdiv_floor(rm, WO);
      int wo = rm - ho * WO;
      xb[it] = X + (long)b * Cin * Hin * Win;
      h0a[it] = ho * stride - pad;
      w0a[it] = wo * stride - pad;
    }
  }

  v8f acc0 = {}, acc1 = {};
  for (int k0 = 0; k0 < Kpad; k0 += 32) {
    if (conv_mode) {
      int k = k0 + lane;
      bool kok = (k < Ktrue);
      int cin = 0, kh = 0, kw = 0;
      if (kok) {
        cin = fdiv_floor(k, khkw);
        int rk = k - cin * khkw;
        kh = fdiv_floor(rk, KW);
        kw = rk - kh * KW;
      }
      long cbase = (long)cin * Hin;
#pragma unroll
      for (int it = 0; it < 8; ++it) {
        float v = 0.f;
        int h = h0a[it] + kh, w = w0a[it] + kw;
        if (kok && mok[it] && (unsigned)h < (unsigned)Hin && (unsigned)w < (unsigned)Win)
          v = xb[it][(cbase + h) * Win + w];
        Alds[wave + 8 * it][lane] = (_Float16)v;
      }
    } else {
      int ml = tid >> 2, kq = (tid & 3) * 8;
      int m = m0 + ml;
      v8h hv = {};
      if (m < M) {
        const float* src = X + (long)m * Kpad + k0 + kq;
#pragma unroll
        for (int j = 0; j < 8; ++j) hv[j] = (_Float16)src[j];
      }
      *(v8h*)&Alds[ml][kq] = hv;
    }
    {
      int nl = tid >> 2, kq = (tid & 3) * 8;
      int n = n0 + nl;
      v8h hv = {};
      if (n < N) hv = *(const v8h*)(W + (long)n * Kpad + k0 + kq);
      *(v8h*)&Blds[nl][kq] = hv;
    }
    __syncthreads();
    int arow = wm * 16 + (lane & 15);
    int koff = (lane < 16) ? 0 : 8;
    v16h af = cat8(*(const v8h*)&Alds[arow][koff], *(const v8h*)&Alds[arow][16 + koff]);
    int kb = (lane < 16) ? 0 : 16;
    int bn = wn2 * 32 + (lane & 15);
    v16h bf0 = cat8(*(const v8h*)&Blds[bn][koff], *(const v8h*)&Blds[bn][16 + koff]);
    v16h bf1 = cat8(*(const v8h*)&Blds[bn + 16][koff], *(const v8h*)&Blds[bn + 16][16 + koff]);
    acc0 = __builtin_amdgcn_wmma_f32_16x16x32_f16(false, af, false, bf0, (short)0, acc0, false, false);
    acc1 = __builtin_amdgcn_wmma_f32_16x16x32_f16(false, af, false, bf1, (short)0, acc1, false, false);
    __syncthreads();
  }
  v8f accs[2] = {acc0, acc1};
  int rowoff = (lane < 16) ? 0 : 8;
#pragma unroll
  for (int half = 0; half < 2; ++half) {
    int n = n0 + wn2 * 32 + half * 16 + (lane & 15);
    if (n >= N) continue;
#pragma unroll
    for (int r = 0; r < 8; ++r) {
      int m = m0 + wm * 16 + rowoff + r;
      if (m >= M) continue;
      float v = accs[half][r];
      if (epi == 1) {
        int b = fdiv_floor(m, howo);
        int rm = m - b * howo;
        v = v * (scale[n] * BN_INVF) + shift[n];
        long oi = ((long)(b * N + n)) * howo + rm;
        if (resid) v += resid[oi];
        if (relu) v = fmaxf(v, 0.f);
        out[oi] = v;
      } else {
        if (shift) v += shift[n];
        if (epi == 2) v = tanhf(v);
        out[(long)m * N + n] = v;
      }
    }
  }
}

/* ---------------- CBAM helpers ------------------------------------------ */
__global__ void k_chan_pool(const float* __restrict__ x, float* avg, float* mx, int HW) {
  int bc = blockIdx.x;
  const float* p = x + (long)bc * HW;
  float s = 0.f, m = -1e30f;
  for (int i = threadIdx.x; i < HW; i += 256) { float v = p[i]; s += v; m = fmaxf(m, v); }
  __shared__ float ss[256], sm[256];
  ss[threadIdx.x] = s; sm[threadIdx.x] = m; __syncthreads();
  for (int o = 128; o > 0; o >>= 1) {
    if (threadIdx.x < o) { ss[threadIdx.x] += ss[threadIdx.x + o]; sm[threadIdx.x] = fmaxf(sm[threadIdx.x], sm[threadIdx.x + o]); }
    __syncthreads();
  }
  if (threadIdx.x == 0) { avg[bc] = ss[0] / HW; mx[bc] = sm[0]; }
}

__global__ void k_cbam_ca(const float* avg, const float* mx, const float* fc1,
                          const float* fc2, float* ca, int C, int Hh) {
  int b = blockIdx.x, tid = threadIdx.x;
  __shared__ float sav[512], smx[512], h1[32], h2[32];
  for (int c = tid; c < C; c += 256) { sav[c] = avg[b * C + c]; smx[c] = mx[b * C + c]; }
  __syncthreads();
  if (tid < Hh) {
    float u = 0.f, v = 0.f;
    for (int c = 0; c < C; ++c) { float w = fc1[tid * C + c]; u += w * sav[c]; v += w * smx[c]; }
    h1[tid] = fmaxf(u, 0.f); h2[tid] = fmaxf(v, 0.f);
  }
  __syncthreads();
  for (int c = tid; c < C; c += 256) {
    float o = 0.f;
    for (int j = 0; j < Hh; ++j) o += fc2[c * Hh + j] * (h1[j] + h2[j]);
    ca[b * C + c] = 1.f / (1.f + expf(-o));
  }
}

__global__ void k_mul_ca(float* x, const float* ca, int HW, long total) {
  long i = (long)blockIdx.x * 256 + threadIdx.x; if (i >= total) return;
  x[i] *= ca[i / HW];
}

__global__ void k_spatial_s(const float* __restrict__ x, float* s, int C, int HW, int tot) {
  int i = blockIdx.x * 256 + threadIdx.x; if (i >= tot) return;
  int b = i / HW, pix = i % HW;
  const float* p = x + ((long)b * C) * HW + pix;
  float sum = 0.f, mx = -1e30f;
  for (int c = 0; c < C; ++c) { float v = p[(long)c * HW]; sum += v; mx = fmaxf(mx, v); }
  s[((long)b * 2) * HW + pix] = sum / C;
  s[((long)b * 2 + 1) * HW + pix] = mx;
}

__global__ void k_sa_conv(const float* __restrict__ s, const float* __restrict__ w,
                          float* sa, int H, int Wd, int tot) {
  int i = blockIdx.x * 256 + threadIdx.x; if (i >= tot) return;
  int hw = H * Wd;
  int b = i / hw, pix = i % hw, h = pix / Wd, x_ = pix % Wd;
  float acc = 0.f;
  for (int c = 0; c < 2; ++c)
    for (int kh = 0; kh < 7; ++kh)
      for (int kw = 0; kw < 7; ++kw) {
        int hh = h - 3 + kh, ww = x_ - 3 + kw;
        if ((unsigned)hh < (unsigned)H && (unsigned)ww < (unsigned)Wd)
          acc += s[((long)(b * 2 + c) * H + hh) * Wd + ww] * w[c * 49 + kh * 7 + kw];
      }
  sa[i] = 1.f / (1.f + expf(-acc));
}

__global__ void k_mul_sa(float* x, const float* sa, int C, int HW, long total) {
  long i = (long)blockIdx.x * 256 + threadIdx.x; if (i >= total) return;
  long b = i / ((long)C * HW); long pix = i % HW;
  x[i] *= sa[b * HW + pix];
}

/* ---------------- FiLM --------------------------------------------------- */
__global__ void k_film_gb(const float* cond, const float* w, const float* bb, float* gb) {
  int b = blockIdx.x;
  for (int j = threadIdx.x; j < 1024; j += 256) {
    float a = bb[j];
    for (int c = 0; c < 64; ++c) a += cond[b * 64 + c] * w[j * 64 + c];
    gb[b * 1024 + j] = a;
  }
}

__global__ void k_film_apply(float* x, const float* gb, int C, int HW, long total) {
  long i = (long)blockIdx.x * 256 + threadIdx.x; if (i >= total) return;
  long b = i / ((long)C * HW); long c = (i / HW) % C;
  x[i] = x[i] * gb[b * 1024 + c] + gb[b * 1024 + 512 + c];
}

/* ---------------- frequency mean -> (B,T2,C) ----------------------------- */
__global__ void k_fmean(const float* __restrict__ x, float* xs, int C, int F, int Wd) {
  int i = blockIdx.x * 256 + threadIdx.x;
  int tot = BB * Wd * C; if (i >= tot) return;
  int b = i / (Wd * C); int rm = i % (Wd * C); int t = rm / C; int c = rm % C;
  float s = 0.f;
  for (int f = 0; f < F; ++f) s += x[(((long)(b * C + c)) * F + f) * Wd + t];
  xs[i] = s / F;
}

/* ---------------- GRU recurrent scan (one block per direction) -----------
 * h(16xGHC) lives in LDS (padded rows, b128 fragment loads); per step
 * hw = h @ whh^T via 48 WMMA tiles across 8 waves; whh fragments are
 * pairs of 16B global loads.                                            */
__global__ __launch_bounds__(256) void k_gru_scan(
    const float* __restrict__ xw0, const float* __restrict__ xw1,
    const _Float16* __restrict__ whh0, const _Float16* __restrict__ whh1,
    const float* __restrict__ bhh0, const float* __restrict__ bhh1,
    float* __restrict__ yout) {
  int dir = blockIdx.x;
  const float* xw = dir ? xw1 : xw0;
  const _Float16* whh = dir ? whh1 : whh0;
  const float* bhh = dir ? bhh1 : bhh0;
  int tid = threadIdx.x, lane = tid & 31, wave = tid >> 5;
  __shared__ alignas(16) _Float16 hh[16][GHC + 8];   /* ~8.3 KB */
  __shared__ float hw[16][3 * GHC];                  /* 48 KB   */
  for (int i = tid; i < 16 * GHC; i += 256) hh[i / GHC][i % GHC] = (_Float16)0.f;
  __syncthreads();
  for (int step = 0; step < T2; ++step) {
    int t = dir ? (T2 - 1 - step) : step;
    for (int tile = wave; tile < 48; tile += 8) {
      v8f acc = {};
      int col = tile * 16 + (lane & 15);
      int kb = (lane < 16) ? 0 : 16;
      int koff = (lane < 16) ? 0 : 8;
      int row = lane & 15;
      for (int k0 = 0; k0 < GHC; k0 += 32) {
        v16h af = cat8(*(const v8h*)&hh[row][k0 + koff], *(const v8h*)&hh[row][k0 + 16 + koff]);
        const v8h* wp = (const v8h*)(whh + (long)col * GHC + k0 + koff);
        v16h bf = cat8(wp[0], wp[2]);
        acc = __builtin_amdgcn_wmma_f32_16x16x32_f16(false, af, false, bf, (short)0, acc, false, false);
      }
      int rowoff = (lane < 16) ? 0 : 8;
#pragma unroll
      for (int r = 0; r < 8; ++r) hw[rowoff + r][col] = acc[r];
    }
    __syncthreads();
    for (int i = tid; i < 16 * GHC; i += 256) {
      int b = i / GHC, c = i % GHC;
      int m = b * T2 + t;
      float xr = xw[(long)m * 768 + c];
      float xz = xw[(long)m * 768 + 256 + c];
      float xn = xw[(long)m * 768 + 512 + c];
      float hr = hw[b][c] + bhh[c];
      float hz = hw[b][256 + c] + bhh[256 + c];
      float hn = hw[b][512 + c] + bhh[512 + c];
      float r = 1.f / (1.f + expf(-(xr + hr)));
      float z = 1.f / (1.f + expf(-(xz + hz)));
      float n = tanhf(xn + r * hn);
      float hp = (float)hh[b][c];
      float h = (1.f - z) * n + z * hp;
      yout[((long)(b * T2 + t)) * 512 + dir * GHC + c] = h;
      hh[b][c] = (_Float16)h;
    }
    __syncthreads();
  }
}

/* ---------------- attention score & final head --------------------------- */
__global__ void k_pool_a(const float* th, const float* w2, const float* b2, float* a) {
  int i = blockIdx.x * 256 + threadIdx.x; if (i >= BB * T2) return;
  float s = b2[0];
  for (int j = 0; j < 128; ++j) s += th[(long)i * 128 + j] * w2[j];
  a[i] = s;
}

__global__ __launch_bounds__(256) void k_final(
    const float* __restrict__ y, const float* __restrict__ a,
    const float* bng, const float* bnb, const float* fcw, const float* fcb,
    float* __restrict__ out) {
  int b = blockIdx.x, tid = threadIdx.x;
  __shared__ float w[T2];
  __shared__ float emb[1024];
  if (tid == 0) {
    float mx = -1e30f;
    for (int t = 0; t < T2; ++t) mx = fmaxf(mx, a[b * T2 + t]);
    float s = 0.f;
    for (int t = 0; t < T2; ++t) { float e = expf(a[b * T2 + t] - mx); w[t] = e; s += e; }
    for (int t = 0; t < T2; ++t) w[t] /= s;
  }
  __syncthreads();
  for (int c = tid; c < 512; c += 256) {
    float mu = 0.f;
    for (int t = 0; t < T2; ++t) mu += y[((long)(b * T2 + t)) * 512 + c] * w[t];
    float var = 0.f;
    for (int t = 0; t < T2; ++t) { float d = y[((long)(b * T2 + t)) * 512 + c] - mu; var += d * d * w[t]; }
    emb[c] = mu; emb[512 + c] = sqrtf(var + 1e-6f);
  }
  __syncthreads();
  for (int d = tid; d < 1024; d += 256) emb[d] = emb[d] * (bng[d] * BN_INVF) + bnb[d];
  __syncthreads();
  if (tid < 10) {
    float o = fcb[tid];
    for (int d = 0; d < 1024; ++d) o += emb[d] * fcw[tid * 1024 + d];
    out[b * 10 + tid] = o;
  }
}

/* ======================= host orchestration ============================== */
extern "C" void kernel_launch(void* const* d_in, const int* in_sizes, int n_in,
                              void* d_out, int out_size, void* d_ws, size_t ws_size,
                              hipStream_t stream) {
  (void)in_sizes; (void)n_in; (void)out_size; (void)ws_size;
  int ci = 0;
  auto NXT = [&]() { return (const float*)d_in[ci++]; };

  /* top-level insertion order, params flattened pytree-style (sorted keys) */
  const float* mel_big = NXT();
  const float* mel_small = NXT();
  const float* cond = NXT();
  const float* bn1_b = NXT(); const float* bn1_g = NXT();
  const float* bno_b = NXT(); const float* bno_g = NXT();
  const float *cb_fc1[4], *cb_fc2[4], *cb_sa[4];
  for (int i = 0; i < 4; ++i) { cb_fc1[i] = NXT(); cb_fc2[i] = NXT(); cb_sa[i] = NXT(); }
  const float* conv1_w = NXT();
  const float* fc_b = NXT(); const float* fc_w = NXT();
  const float* film_b = NXT(); const float* film_w = NXT();
  const float *g_bhh[2][2], *g_bih[2][2], *g_whh[2][2], *g_wih[2][2]; /* [layer][0='b',1='f'] */
  for (int l = 0; l < 2; ++l)
    for (int d = 0; d < 2; ++d) { g_bhh[l][d] = NXT(); g_bih[l][d] = NXT(); g_whh[l][d] = NXT(); g_wih[l][d] = NXT(); }
  const float* ibn_b = NXT(); const float* ibn_g = NXT();
  struct Blk { const float *bn1b,*bn1g,*bn2b,*bn2g,*dbnb,*dbng,*dw,*w1,*w2; int hasdw; };
  Blk blk[16]; int nb = 0;
  int cfg[4][4] = {{64,64,3,1},{64,128,4,2},{128,256,6,2},{256,512,3,2}};
  for (int l = 0; l < 4; ++l)
    for (int bi = 0; bi < cfg[l][2]; ++bi) {
      Blk t{}; t.hasdw = (bi == 0) && (cfg[l][3] != 1 || cfg[l][0] != cfg[l][1]);
      t.bn1b = NXT(); t.bn1g = NXT(); t.bn2b = NXT(); t.bn2g = NXT();
      if (t.hasdw) { t.dbnb = NXT(); t.dbng = NXT(); t.dw = NXT(); }
      t.w1 = NXT(); t.w2 = NXT();
      blk[nb++] = t;
    }
  const float* p0a = NXT(); const float* p0d = NXT(); const float* p0r = NXT();
  const float* p1a = NXT(); const float* p1d = NXT(); const float* p1r = NXT();
  const float* pool_b1 = NXT(); const float* pool_b2 = NXT();
  const float* pool_w1 = NXT(); const float* pool_w2 = NXT();

  /* bump allocator over d_ws */
  char* base = (char*)d_ws; size_t off = 0;
  auto allocF = [&](size_t n) -> float* { float* p = (float*)(base + off); off = (off + n * 4 + 255) & ~(size_t)255; return p; };
  auto allocH = [&](size_t n) -> _Float16* { _Float16* p = (_Float16*)(base + off); off = (off + n * 2 + 255) & ~(size_t)255; return p; };

  auto w2h = [&](const float* src, int N, int Kt) -> _Float16* {
    int Kp = (Kt + 31) & ~31;
    _Float16* dst = allocH((size_t)N * Kp);
    long tot = (long)N * Kp;
    k_w2h<<<dim3((unsigned)((tot + 255) / 256)), 256, 0, stream>>>(src, dst, N, Kt, Kp);
    return dst;
  };
  auto conv = [&](const float* x, const _Float16* wf, float* out_,
                  int Cin, int Hin, int Win, int Cout, int K, int stride, int pad,
                  const float* g, const float* bbn, const float* resid, int relu) {
    int HO = (Hin + 2 * pad - K) / stride + 1;
    int WO = (Win + 2 * pad - K) / stride + 1;
    int M = BB * HO * WO; int Kt = Cin * K * K; int Kp = (Kt + 31) & ~31;
    dim3 grid((unsigned)((M + 63) / 64), (unsigned)((Cout + 63) / 64));
    k_wmma_gemm<<<grid, 256, 0, stream>>>(x, wf, out_, M, Cout, Kt, Kp,
        1, Cin, Hin, Win, K, K, stride, pad, HO, WO, 1, g, bbn, resid, relu);
  };
  auto gemm = [&](const float* x, const _Float16* wf, float* out_, int M, int N, int K,
                  const float* bias, int epi) {
    dim3 grid((unsigned)((M + 63) / 64), (unsigned)((N + 63) / 64));
    k_wmma_gemm<<<grid, 256, 0, stream>>>(x, wf, out_, M, N, K, K,
        0, 0, 0, 0, 0, 0, 0, 0, 1, 1, epi, nullptr, bias, nullptr, 0);
  };

  /* weight conversions */
  _Float16* wf_conv1 = w2h(conv1_w, 64, 2 * 49);
  _Float16 *wf1[16], *wf2[16], *wfd[16];
  {
    int idx = 0;
    for (int l = 0; l < 4; ++l)
      for (int bi = 0; bi < cfg[l][2]; ++bi) {
        int cin = (bi == 0) ? cfg[l][0] : cfg[l][1];
        int cout = cfg[l][1];
        wf1[idx] = w2h(blk[idx].w1, cout, cin * 9);
        wf2[idx] = w2h(blk[idx].w2, cout, cout * 9);
        wfd[idx] = blk[idx].hasdw ? w2h(blk[idx].dw, cout, cin) : nullptr;
        ++idx;
      }
  }
  _Float16 *wf_wih[2][2], *wf_whh[2][2];
  for (int l = 0; l < 2; ++l)
    for (int d = 0; d < 2; ++d) { wf_wih[l][d] = w2h(g_wih[l][d], 768, 512); wf_whh[l][d] = w2h(g_whh[l][d], 768, 256); }
  _Float16* wf_pw1 = w2h(pool_w1, 128, 512);

  /* activation buffers */
  float* spec = allocF((size_t)BB * 2 * NMELS * TT);
  size_t bigN = (size_t)BB * 64 * 64 * 256;
  float* bufA = allocF(bigN); float* bufB = allocF(bigN); float* bufC = allocF(bigN);
  float* bufD = allocF((size_t)BB * 128 * 32 * 128);
  float* avg = allocF(BB * 512); float* mxp = allocF(BB * 512); float* ca = allocF(BB * 512);
  float* sbuf = allocF((size_t)BB * 2 * 64 * 256);
  float* sabuf = allocF((size_t)BB * 64 * 256);
  float* gb = allocF(BB * 1024);
  float* xseq = allocF((size_t)BB * T2 * 512);
  float* y1 = allocF((size_t)BB * T2 * 512);
  float* y2 = allocF((size_t)BB * T2 * 512);
  float* xwf = allocF((size_t)BB * T2 * 768);
  float* xwb = allocF((size_t)BB * T2 * 768);
  float* th = allocF((size_t)BB * T2 * 128);
  float* abuf = allocF(BB * T2);

  /* 1. PCEN + input BN */
  k_pcen<<<(2 * BB * NMELS + 255) / 256, 256, 0, stream>>>(
      mel_big, mel_small, p0a, p0d, p0r, p1a, p1d, p1r, ibn_b, ibn_g, spec);

  /* 2. stem conv */
  conv(spec, wf_conv1, bufA, 2, 128, 512, 64, 7, 2, 3, bn1_g, bn1_b, nullptr, 1);

  /* 3. ResNet layers + CBAM */
  float* x = bufA; float* t1 = bufB; float* t2 = bufC;
  int C = 64, H = 64, Wd = 256, bidx = 0;
  for (int l = 0; l < 4; ++l) {
    int Cout = cfg[l][1];
    for (int bi = 0; bi < cfg[l][2]; ++bi) {
      Blk& pb = blk[bidx];
      int s = (bi == 0) ? cfg[l][3] : 1;
      int HO = (H - 1) / s + 1, WO = (Wd - 1) / s + 1;
      const float* resid;
      if (pb.hasdw) {
        conv(x, wfd[bidx], bufD, C, H, Wd, Cout, 1, s, 0, pb.dbng, pb.dbnb, nullptr, 0);
        resid = bufD;
      } else resid = x;
      conv(x, wf1[bidx], t1, C, H, Wd, Cout, 3, s, 1, pb.bn1g, pb.bn1b, nullptr, 1);
      conv(t1, wf2[bidx], t2, Cout, HO, WO, Cout, 3, 1, 1, pb.bn2g, pb.bn2b, resid, 1);
      float* tmp = x; x = t2; t2 = tmp;
      C = Cout; H = HO; Wd = WO; ++bidx;
    }
    int HW = H * Wd; long total = (long)BB * C * HW;
    int hdim = (C / 16 < 4) ? 4 : C / 16;
    k_chan_pool<<<BB * C, 256, 0, stream>>>(x, avg, mxp, HW);
    k_cbam_ca<<<BB, 256, 0, stream>>>(avg, mxp, cb_fc1[l], cb_fc2[l], ca, C, hdim);
    k_mul_ca<<<(unsigned)((total + 255) / 256), 256, 0, stream>>>(x, ca, HW, total);
    k_spatial_s<<<(BB * HW + 255) / 256, 256, 0, stream>>>(x, sbuf, C, HW, BB * HW);
    k_sa_conv<<<(BB * HW + 255) / 256, 256, 0, stream>>>(sbuf, cb_sa[l], sabuf, H, Wd, BB * HW);
    k_mul_sa<<<(unsigned)((total + 255) / 256), 256, 0, stream>>>(x, sabuf, C, HW, total);
  }

  /* 4. FiLM */
  k_film_gb<<<BB, 256, 0, stream>>>(cond, film_w, film_b, gb);
  { long total = (long)BB * C * H * Wd;
    k_film_apply<<<(unsigned)((total + 255) / 256), 256, 0, stream>>>(x, gb, C, H * Wd, total); }

  /* 5. frequency mean -> (B, T2, 512) */
  { int tot = BB * Wd * C;
    k_fmean<<<(tot + 255) / 256, 256, 0, stream>>>(x, xseq, C, H, Wd); }

  /* 6. GRU layer 1 (dir index: [l][0]='b' backward, [l][1]='f' forward) */
  gemm(xseq, wf_wih[0][1], xwf, BB * T2, 768, 512, g_bih[0][1], 0);
  gemm(xseq, wf_wih[0][0], xwb, BB * T2, 768, 512, g_bih[0][0], 0);
  k_gru_scan<<<2, 256, 0, stream>>>(xwf, xwb, wf_whh[0][1], wf_whh[0][0],
                                    g_bhh[0][1], g_bhh[0][0], y1);
  /* GRU layer 2 */
  gemm(y1, wf_wih[1][1], xwf, BB * T2, 768, 512, g_bih[1][1], 0);
  gemm(y1, wf_wih[1][0], xwb, BB * T2, 768, 512, g_bih[1][0], 0);
  k_gru_scan<<<2, 256, 0, stream>>>(xwf, xwb, wf_whh[1][1], wf_whh[1][0],
                                    g_bhh[1][1], g_bhh[1][0], y2);

  /* 7. attentive pooling + BN + FC */
  gemm(y2, wf_pw1, th, BB * T2, 128, 512, pool_b1, 2);
  k_pool_a<<<(BB * T2 + 255) / 256, 256, 0, stream>>>(th, pool_w2, pool_b2, abuf);
  k_final<<<BB, 256, 0, stream>>>(y2, abuf, bno_g, bno_b, fc_w, fc_b, (float*)d_out);
}
